// SelfAttention2_35424890258013
// MI455X (gfx1250) — matrix operand order repaired and checked by reading, not compiled, not run on hardware
//
#include <hip/hip_runtime.h>
#include <hip/hip_bf16.h>

// ---------------------------------------------------------------------------
// Self-attention (B=4, S=2048, H=512, 8 heads, d=64), fp32 in/out.
// All matmuls via v_wmma_f32_16x16x32_bf16 (fp32 accumulate).
// Workspace requirement: ~66 MB.
// ---------------------------------------------------------------------------

typedef __bf16 v16bf __attribute__((ext_vector_type(16)));
typedef float  v8f   __attribute__((ext_vector_type(8)));

#define B_    4
#define S_    2048
#define H_    512
#define NH    8
#define DH    64
#define N3H   1536
#define MT    512            // (B*S)/16 M-tiles
#define KT_H  16             // H/32 K-tiles
#define NT3   96             // 3H/16 N-tiles
#define NTH   32             // H/16 N-tiles
#define FRAG_H 512           // halfs per fragment (32 lanes * 16)
#define NEG_BIG (-1.0e30f)

union Frag { uint4 u4[2]; v16bf v; unsigned short s[16]; };
union AccU { v8f v; float f[8]; };

__device__ __forceinline__ unsigned short f2bf(float f) {
  unsigned int u = __float_as_uint(f);
  u += 0x7FFFu + ((u >> 16) & 1u);          // round-to-nearest-even
  return (unsigned short)(u >> 16);
}
// A-matrix 16x32 bf16 register mapping (ISA 7.12.2): element e of lane
// (half = lane/16) holds K = 8*half + e (+8 when e >= 8).
__device__ __forceinline__ int kmapA(int e, int half) {
  return 8 * half + e + ((e >= 8) ? 8 : 0);
}
// B-matrix 32x16 bf16 mapping: lane holds column N=lane%16,
// element e holds K = 8*half + e (+8 when e >= 8), same as A.

#define WMMA_BF16(A, Bm, C) \
  __builtin_amdgcn_wmma_f32_16x16x32_bf16(false, (A), false, (Bm), (short)0, (C), false, false)

// ---------------------------------------------------------------------------
// Pack x (fp32 [B*S, H]) -> A-fragments [mt=512][kt=16][lane][16] bf16
// ---------------------------------------------------------------------------
__global__ void pack_x_kernel(const float* __restrict__ x,
                              unsigned short* __restrict__ Ap) {
  int tid = blockIdx.x * blockDim.x + threadIdx.x;
  int lane = tid & 31, f = tid >> 5;
  if (f >= MT * KT_H) return;
  int kt = f % KT_H, mt = f / KT_H;
  int half = lane >> 4;
  int row = mt * 16 + (lane & 15);
  const float* src = x + (size_t)row * H_ + kt * 32;
  Frag fr;
#pragma unroll
  for (int e = 0; e < 16; ++e) fr.s[e] = f2bf(src[kmapA(e, half)]);
  uint4* dst = (uint4*)(Ap + (size_t)f * FRAG_H + lane * 16);
  dst[0] = fr.u4[0]; dst[1] = fr.u4[1];
}

// ---------------------------------------------------------------------------
// Pack weight (fp32 [K=rows, N=ncols]) -> B-fragments [nt][kt][lane][16] bf16
// ---------------------------------------------------------------------------
__global__ void pack_w_kernel(const float* __restrict__ W,
                              unsigned short* __restrict__ Bp,
                              int ncols, int nfrag, int kt_count) {
  int tid = blockIdx.x * blockDim.x + threadIdx.x;
  int lane = tid & 31, f = tid >> 5;
  if (f >= nfrag) return;
  int kt = f % kt_count, nt = f / kt_count;
  int half = lane >> 4;
  int col = nt * 16 + (lane & 15);
  Frag fr;
#pragma unroll
  for (int e = 0; e < 16; ++e) {
    int k = kt * 32 + 8 * half + e + ((e >= 8) ? 8 : 0);
    fr.s[e] = f2bf(W[(size_t)k * ncols + col]);
  }
  uint4* dst = (uint4*)(Bp + (size_t)f * FRAG_H + lane * 16);
  dst[0] = fr.u4[0]; dst[1] = fr.u4[1];
}

// ---------------------------------------------------------------------------
// QKV GEMM: [8192 x 1536] = A[8192 x 512] * W[512 x 1536] + b.
// 8 waves/block; wave computes 64x32 (4 mtiles x 2 ntiles).
// Q part (cols < 512) scaled by 1/sqrt(d)=0.125 at store. Output bf16.
// ---------------------------------------------------------------------------
__global__ void __launch_bounds__(256) qkv_gemm_kernel(
    const unsigned short* __restrict__ Ap, const unsigned short* __restrict__ Bp,
    const float* __restrict__ bias, unsigned short* __restrict__ qkv) {
  int tid = threadIdx.x, lane = tid & 31, wid = tid >> 5;
  int half = lane >> 4;
  int waveM = wid >> 2, waveN = wid & 3;
  int mtBase = blockIdx.x * 8 + waveM * 4;
  int ntBase = blockIdx.y * 8 + waveN * 2;
  AccU acc[4][2];
#pragma unroll
  for (int mi = 0; mi < 4; ++mi)
#pragma unroll
    for (int ni = 0; ni < 2; ++ni)
#pragma unroll
      for (int r = 0; r < 8; ++r) acc[mi][ni].f[r] = 0.0f;

  for (int kt = 0; kt < KT_H; ++kt) {
    if (kt + 1 < KT_H) {
      __builtin_prefetch(Ap + ((size_t)(mtBase * KT_H + kt + 1)) * FRAG_H + lane * 16, 0, 3);
      __builtin_prefetch(Bp + ((size_t)(ntBase * KT_H + kt + 1)) * FRAG_H + lane * 16, 0, 3);
    }
    Frag a[4], b[2];
#pragma unroll
    for (int mi = 0; mi < 4; ++mi) {
      const uint4* p = (const uint4*)(Ap + ((size_t)((mtBase + mi) * KT_H + kt)) * FRAG_H + lane * 16);
      a[mi].u4[0] = p[0]; a[mi].u4[1] = p[1];
    }
#pragma unroll
    for (int ni = 0; ni < 2; ++ni) {
      const uint4* p = (const uint4*)(Bp + ((size_t)((ntBase + ni) * KT_H + kt)) * FRAG_H + lane * 16);
      b[ni].u4[0] = p[0]; b[ni].u4[1] = p[1];
    }
#pragma unroll
    for (int mi = 0; mi < 4; ++mi)
#pragma unroll
      for (int ni = 0; ni < 2; ++ni)
        acc[mi][ni].v = WMMA_BF16(a[mi].v, b[ni].v, acc[mi][ni].v);
  }
#pragma unroll
  for (int ni = 0; ni < 2; ++ni) {
    int col = (ntBase + ni) * 16 + (lane & 15);
    float bv = bias[col];
    float qs = (col < H_) ? 0.125f : 1.0f;   // 1/sqrt(64) applied to Q
#pragma unroll
    for (int mi = 0; mi < 4; ++mi)
#pragma unroll
      for (int r = 0; r < 8; ++r) {
        int row = (mtBase + mi) * 16 + r + 8 * half;   // C-layout: M = r + 8*half
        qkv[(size_t)row * N3H + col] = f2bf((acc[mi][ni].f[r] + bv) * qs);
      }
  }
}

// ---------------------------------------------------------------------------
// Repack Q -> A-fragments [bh=32][sqt=128][dkt=2][lane][16]
// ---------------------------------------------------------------------------
__global__ void pack_q_kernel(const unsigned short* __restrict__ qkv,
                              unsigned short* __restrict__ Qp) {
  int tid = blockIdx.x * blockDim.x + threadIdx.x;
  int lane = tid & 31, f = tid >> 5;
  if (f >= B_ * NH * (S_ / 16) * 2) return;
  int dkt = f & 1, r1 = f >> 1;
  int sqt = r1 & 127, bh = r1 >> 7;
  int h = bh & 7, b = bh >> 3;
  int half = lane >> 4;
  int row = b * S_ + sqt * 16 + (lane & 15);
  const unsigned short* src = qkv + (size_t)row * N3H + h * DH + dkt * 32;
  Frag fr;
#pragma unroll
  for (int e = 0; e < 16; ++e) fr.s[e] = src[kmapA(e, half)];
  uint4* dst = (uint4*)(Qp + (size_t)f * FRAG_H + lane * 16);
  dst[0] = fr.u4[0]; dst[1] = fr.u4[1];
}

// ---------------------------------------------------------------------------
// Repack K -> B-fragments (B = K^T: n=key, k=dcomp), contiguous copy.
// [bh][skt=128][dkt=2][lane][16]
// ---------------------------------------------------------------------------
__global__ void pack_k_kernel(const unsigned short* __restrict__ qkv,
                              unsigned short* __restrict__ Kp) {
  int tid = blockIdx.x * blockDim.x + threadIdx.x;
  int lane = tid & 31, f = tid >> 5;
  if (f >= B_ * NH * (S_ / 16) * 2) return;
  int dkt = f & 1, r1 = f >> 1;
  int skt = r1 & 127, bh = r1 >> 7;
  int h = bh & 7, b = bh >> 3;
  int half = lane >> 4;
  int row = b * S_ + skt * 16 + (lane & 15);           // key index
  const uint4* src = (const uint4*)(qkv + (size_t)row * N3H + H_ + h * DH + dkt * 32);
  Frag fr;
  fr.u4[0] = src[half]; fr.u4[1] = src[2 + half];
  uint4* dst = (uint4*)(Kp + (size_t)f * FRAG_H + lane * 16);
  dst[0] = fr.u4[0]; dst[1] = fr.u4[1];
}

// ---------------------------------------------------------------------------
// Repack V -> B-fragments (n=dcomp, k=key). [bh][kt32=64][nt=4][lane][16]
// ---------------------------------------------------------------------------
__global__ void pack_v_kernel(const unsigned short* __restrict__ qkv,
                              unsigned short* __restrict__ Vp) {
  int tid = blockIdx.x * blockDim.x + threadIdx.x;
  int lane = tid & 31, f = tid >> 5;
  if (f >= B_ * NH * (S_ / 32) * 4) return;
  int nt = f & 3, r1 = f >> 2;
  int kt32 = r1 & 63, bh = r1 >> 6;
  int h = bh & 7, b = bh >> 3;
  int half = lane >> 4;
  int col = 2 * H_ + h * DH + nt * 16 + (lane & 15);
  Frag fr;
#pragma unroll
  for (int e = 0; e < 16; ++e) {
    int row = b * S_ + kt32 * 32 + 8 * half + e + ((e >= 8) ? 8 : 0);      // key index
    fr.s[e] = qkv[(size_t)row * N3H + col];
  }
  uint4* dst = (uint4*)(Vp + (size_t)f * FRAG_H + lane * 16);
  dst[0] = fr.u4[0]; dst[1] = fr.u4[1];
}

// ---------------------------------------------------------------------------
// Flash attention: 1 wave per 16-row Q tile, stream 32-key blocks.
// scores via WMMA, fp32 online softmax, P->A-frag via LDS, PV via WMMA.
// Writes O as A-fragments for the output GEMM: Op[mtG=512][kt=16][lane][16]
// ---------------------------------------------------------------------------
__global__ void __launch_bounds__(128) attn_kernel(
    const unsigned short* __restrict__ Qp, const unsigned short* __restrict__ Kp,
    const unsigned short* __restrict__ Vp, const int* __restrict__ mask,
    unsigned short* __restrict__ Op) {
  __shared__ __align__(16) unsigned short ldsP[4][16 * 32];
  __shared__ __align__(16) unsigned short ldsO[4][16 * 64];
  int tid = threadIdx.x, lane = tid & 31, wid = tid >> 5, half = lane >> 4;
  int gw = blockIdx.x * 4 + wid;                 // 0..4095
  int sqt = gw & 127, bh = gw >> 7;              // bh = b*8 + h
  int b = bh >> 3, h = bh & 7;
  int key0 = lane & 15;
  const int* mrow = mask + b * S_;

  Frag qa[2];
#pragma unroll
  for (int dkt = 0; dkt < 2; ++dkt) {
    const uint4* p = (const uint4*)(Qp + (((size_t)bh * 128 + sqt) * 2 + dkt) * FRAG_H + lane * 16);
    qa[dkt].u4[0] = p[0]; qa[dkt].u4[1] = p[1];
  }

  float m[8], l[8];
  AccU o[4];
#pragma unroll
  for (int r = 0; r < 8; ++r) { m[r] = -3.0e38f; l[r] = 0.0f; }
#pragma unroll
  for (int nt = 0; nt < 4; ++nt)
#pragma unroll
    for (int r = 0; r < 8; ++r) o[nt].f[r] = 0.0f;

  for (int kb = 0; kb < S_ / 32; ++kb) {
    float st[2][8];
    int valid[2];
#pragma unroll
    for (int t = 0; t < 2; ++t) {
      int skt = kb * 2 + t;
      Frag kf0, kf1;
      const uint4* p0 = (const uint4*)(Kp + (((size_t)bh * 128 + skt) * 2 + 0) * FRAG_H + lane * 16);
      const uint4* p1 = (const uint4*)(Kp + (((size_t)bh * 128 + skt) * 2 + 1) * FRAG_H + lane * 16);
      kf0.u4[0] = p0[0]; kf0.u4[1] = p0[1];
      kf1.u4[0] = p1[0]; kf1.u4[1] = p1[1];
      AccU s;
#pragma unroll
      for (int r = 0; r < 8; ++r) s.f[r] = 0.0f;
      s.v = WMMA_BF16(qa[0].v, kf0.v, s.v);
      s.v = WMMA_BF16(qa[1].v, kf1.v, s.v);
      valid[t] = mrow[skt * 16 + key0];
#pragma unroll
      for (int r = 0; r < 8; ++r) st[t][r] = valid[t] ? s.f[r] : NEG_BIG;
    }
    // online softmax: row reductions across the 16-lane C-layout group
    float corr[8];
#pragma unroll
    for (int r = 0; r < 8; ++r) {
      float mx = fmaxf(st[0][r], st[1][r]);
      mx = fmaxf(mx, __shfl_xor(mx, 1, 32));
      mx = fmaxf(mx, __shfl_xor(mx, 2, 32));
      mx = fmaxf(mx, __shfl_xor(mx, 4, 32));
      mx = fmaxf(mx, __shfl_xor(mx, 8, 32));
      float mn = fmaxf(m[r], mx);
      corr[r] = __expf(m[r] - mn);
      m[r] = mn;
    }
    float p0[8], p1[8];
#pragma unroll
    for (int r = 0; r < 8; ++r) {
      p0[r] = valid[0] ? __expf(st[0][r] - m[r]) : 0.0f;
      p1[r] = valid[1] ? __expf(st[1][r] - m[r]) : 0.0f;
      float rs = p0[r] + p1[r];
      rs += __shfl_xor(rs, 1, 32);
      rs += __shfl_xor(rs, 2, 32);
      rs += __shfl_xor(rs, 4, 32);
      rs += __shfl_xor(rs, 8, 32);
      l[r] = l[r] * corr[r] + rs;
    }
#pragma unroll
    for (int nt = 0; nt < 4; ++nt)
#pragma unroll
      for (int r = 0; r < 8; ++r) o[nt].f[r] *= corr[r];

    // P (C-layout) -> LDS row-major [16 rows x 32 key cols] bf16
    unsigned short* Pb = ldsP[wid];
#pragma unroll
    for (int r = 0; r < 8; ++r) {
      int rowM = r + 8 * half;
      Pb[rowM * 32 + key0]      = f2bf(p0[r]);
      Pb[rowM * 32 + 16 + key0] = f2bf(p1[r]);
    }
    __syncthreads();
    // read back in A-fragment layout
    Frag pa;
    int prow = lane & 15;
    pa.u4[0] = *(const uint4*)(ldsP[wid] + prow * 32 + 8 * half);
    pa.u4[1] = *(const uint4*)(ldsP[wid] + prow * 32 + 16 + 8 * half);
    __syncthreads();
#pragma unroll
    for (int nt = 0; nt < 4; ++nt) {
      Frag vf;
      const uint4* p = (const uint4*)(Vp + (((size_t)bh * 64 + kb) * 4 + nt) * FRAG_H + lane * 16);
      vf.u4[0] = p[0]; vf.u4[1] = p[1];
      o[nt].v = WMMA_BF16(pa.v, vf.v, o[nt].v);
    }
  }

  // normalize and repack O through LDS into A-fragments for out GEMM
  float inv[8];
#pragma unroll
  for (int r = 0; r < 8; ++r) inv[r] = 1.0f / l[r];
  unsigned short* Ob = ldsO[wid];
#pragma unroll
  for (int nt = 0; nt < 4; ++nt)
#pragma unroll
    for (int r = 0; r < 8; ++r) {
      int rowM = r + 8 * half;
      Ob[rowM * 64 + nt * 16 + key0] = f2bf(o[nt].f[r] * inv[r]);
    }
  __syncthreads();
  int mtG = b * 128 + sqt;
  int orow = lane & 15;
#pragma unroll
  for (int dkt = 0; dkt < 2; ++dkt) {
    Frag of;
    of.u4[0] = *(const uint4*)(Ob + orow * 64 + dkt * 32 + 8 * half);
    of.u4[1] = *(const uint4*)(Ob + orow * 64 + dkt * 32 + 16 + 8 * half);
    uint4* dst = (uint4*)(Op + ((size_t)mtG * 16 + h * 2 + dkt) * FRAG_H + lane * 16);
    dst[0] = of.u4[0]; dst[1] = of.u4[1];
  }
}

// ---------------------------------------------------------------------------
// Output GEMM: out[8192 x 512] = O[8192 x 512] * W_out[512 x 512] + b_out (fp32)
// ---------------------------------------------------------------------------
__global__ void __launch_bounds__(256) out_gemm_kernel(
    const unsigned short* __restrict__ Ap, const unsigned short* __restrict__ Bp,
    const float* __restrict__ bias, float* __restrict__ out) {
  int tid = threadIdx.x, lane = tid & 31, wid = tid >> 5;
  int half = lane >> 4;
  int waveM = wid >> 2, waveN = wid & 3;
  int mtBase = blockIdx.x * 8 + waveM * 4;
  int ntBase = blockIdx.y * 8 + waveN * 2;
  AccU acc[4][2];
#pragma unroll
  for (int mi = 0; mi < 4; ++mi)
#pragma unroll
    for (int ni = 0; ni < 2; ++ni)
#pragma unroll
      for (int r = 0; r < 8; ++r) acc[mi][ni].f[r] = 0.0f;

  for (int kt = 0; kt < KT_H; ++kt) {
    if (kt + 1 < KT_H) {
      __builtin_prefetch(Ap + ((size_t)(mtBase * KT_H + kt + 1)) * FRAG_H + lane * 16, 0, 3);
      __builtin_prefetch(Bp + ((size_t)(ntBase * KT_H + kt + 1)) * FRAG_H + lane * 16, 0, 3);
    }
    Frag a[4], b[2];
#pragma unroll
    for (int mi = 0; mi < 4; ++mi) {
      const uint4* p = (const uint4*)(Ap + ((size_t)((mtBase + mi) * KT_H + kt)) * FRAG_H + lane * 16);
      a[mi].u4[0] = p[0]; a[mi].u4[1] = p[1];
    }
#pragma unroll
    for (int ni = 0; ni < 2; ++ni) {
      const uint4* p = (const uint4*)(Bp + ((size_t)((ntBase + ni) * KT_H + kt)) * FRAG_H + lane * 16);
      b[ni].u4[0] = p[0]; b[ni].u4[1] = p[1];
    }
#pragma unroll
    for (int mi = 0; mi < 4; ++mi)
#pragma unroll
      for (int ni = 0; ni < 2; ++ni)
        acc[mi][ni].v = WMMA_BF16(a[mi].v, b[ni].v, acc[mi][ni].v);
  }
#pragma unroll
  for (int ni = 0; ni < 2; ++ni) {
    int col = (ntBase + ni) * 16 + (lane & 15);
    float bv = bias[col];
#pragma unroll
    for (int mi = 0; mi < 4; ++mi)
#pragma unroll
      for (int r = 0; r < 8; ++r) {
        int row = (mtBase + mi) * 16 + r + 8 * half;
        out[(size_t)row * H_ + col] = acc[mi][ni].f[r] + bv;
      }
  }
}

// ---------------------------------------------------------------------------
extern "C" void kernel_launch(void* const* d_in, const int* in_sizes, int n_in,
                              void* d_out, int out_size, void* d_ws, size_t ws_size,
                              hipStream_t stream) {
  const float* x      = (const float*)d_in[0];
  const int*   mask   = (const int*)d_in[1];
  const float* W_comb = (const float*)d_in[2];
  const float* b_comb = (const float*)d_in[3];
  const float* W_out  = (const float*)d_in[4];
  const float* b_out  = (const float*)d_in[5];
  float* out = (float*)d_out;

  char* w = (char*)d_ws;
  unsigned short* Ap   = (unsigned short*)w; w += (size_t)MT * KT_H * FRAG_H * 2;   //  8 MB
  unsigned short* Bc   = (unsigned short*)w; w += (size_t)NT3 * KT_H * FRAG_H * 2;  //  1.5 MB
  unsigned short* Bo   = (unsigned short*)w; w += (size_t)NTH * KT_H * FRAG_H * 2;  //  0.5 MB
  unsigned short* qkvb = (unsigned short*)w; w += (size_t)B_ * S_ * N3H * 2;        // 24 MB
  unsigned short* Qp   = (unsigned short*)w; w += (size_t)B_ * S_ * H_ * 2;         //  8 MB
  unsigned short* Kp   = (unsigned short*)w; w += (size_t)B_ * S_ * H_ * 2;         //  8 MB
  unsigned short* Vp   = (unsigned short*)w; w += (size_t)B_ * S_ * H_ * 2;         //  8 MB
  unsigned short* Opk  = (unsigned short*)w;                                        //  8 MB

  pack_x_kernel<<<dim3(1024), 256, 0, stream>>>(x, Ap);
  pack_w_kernel<<<dim3(192), 256, 0, stream>>>(W_comb, Bc, N3H, NT3 * KT_H, KT_H);
  pack_w_kernel<<<dim3(64), 256, 0, stream>>>(W_out, Bo, H_, NTH * KT_H, KT_H);
  qkv_gemm_kernel<<<dim3(64, 12), 256, 0, stream>>>(Ap, Bc, b_comb, qkvb);
  pack_q_kernel<<<dim3(1024), 256, 0, stream>>>(qkvb, Qp);
  pack_k_kernel<<<dim3(1024), 256, 0, stream>>>(qkvb, Kp);
  pack_v_kernel<<<dim3(1024), 256, 0, stream>>>(qkvb, Vp);
  attn_kernel<<<dim3(1024), 128, 0, stream>>>(Qp, Kp, Vp, mask, Opk);
  out_gemm_kernel<<<dim3(64, 4), 256, 0, stream>>>(Opk, Bo, b_out, out);
}
